// RWKV_TimeMix_3040836846333
// MI455X (gfx1250) — compile-verified, matrix operand order checked by reading, not run on hardware
//
#include <hip/hip_runtime.h>
#include <hip/hip_bf16.h>

// ---------------------------------------------------------------------------
// RWKV-6 TimeMix for MI455X (gfx1250, wave32, WMMA bf16 16x16x32)
// B=2 T=1024 C=2048 H=32 N=64, M=B*T=2048
// ---------------------------------------------------------------------------

typedef __bf16 bf16_t;
typedef __attribute__((ext_vector_type(16))) __bf16 v16bf;
typedef __attribute__((ext_vector_type(8)))  __bf16 v8bf;
typedef __attribute__((ext_vector_type(8)))  float  v8f;

#define RWKV_B 2
#define RWKV_T 1024
#define RWKV_C 2048
#define RWKV_H 32
#define RWKV_N 64
#define RWKV_M (RWKV_B * RWKV_T)
#define GN_EPS 6.4e-4f   // 1e-5 * 8^2

// ---- CDNA5 async global->LDS path (guarded: falls back to sync LDS store) --
#ifndef HAVE_ASYNC_LDS
#if defined(__has_builtin)
#if __has_builtin(__builtin_amdgcn_global_load_async_to_lds_b32) && \
    __has_builtin(__builtin_amdgcn_s_wait_asynccnt)
#define HAVE_ASYNC_LDS 1
#endif
#endif
#endif
#ifndef HAVE_ASYNC_LDS
#define HAVE_ASYNC_LDS 0
#endif

#if HAVE_ASYNC_LDS
typedef __attribute__((address_space(3))) int lds_int_t;
typedef __attribute__((address_space(1))) int glb_int_t;
// flat->LDS addrspacecast == 32-bit truncation on AMDGPU; flat->global is
// numerically identity, so integer round-trips are safe here.
#define AS3_PTR(p) ((lds_int_t*)(unsigned int)(unsigned long long)(p))
#define AS1_PTR(p) ((glb_int_t*)(unsigned long long)(p))
#endif

// ---------------------------------------------------------------------------
// Fragment loader: 16-bit A/B matrix layout for V_WMMA_*_16X16X32 (wave32).
// Lane L (L<16): row = L, holds K = {0..7} then {16..23}.
// Lane L (L>=16): row = L-16, holds K = {8..15} then {24..31}.
// With a row-major [16 x K] source (B supplied pre-transposed as [N,K]),
// each lane does two contiguous 16-byte loads.
// ---------------------------------------------------------------------------
__device__ __forceinline__ v16bf frag_load(const bf16_t* __restrict__ base, int ld, int lane) {
  const int m  = lane & 15;
  const int kb = (lane >> 4) << 3;
  const bf16_t* p = base + (size_t)m * ld + kb;
  v8bf h0 = *(const v8bf*)(p);
  v8bf h1 = *(const v8bf*)(p + 16);
  v16bf a;
#pragma unroll
  for (int i = 0; i < 8; ++i) { a[i] = h0[i]; a[i + 8] = h1[i]; }
  return a;
}

// ---------------------------------------------------------------------------
// Weight convert + transpose: src f32 [K,N] -> dst bf16 [N,K]
// ---------------------------------------------------------------------------
__global__ void k_transpose_bf16(const float* __restrict__ S, bf16_t* __restrict__ D,
                                 int K, int N) {
  size_t idx = (size_t)blockIdx.x * blockDim.x + threadIdx.x;
  if (idx >= (size_t)K * N) return;
  int k = (int)(idx / N);
  int n = (int)(idx % N);
  D[(size_t)n * K + k] = (bf16_t)S[idx];
}

// ---------------------------------------------------------------------------
// Token shift: xx = x[t-1]-x[t] (0 at t==0), Xx = bf16(x + xx*maa_x)
// ---------------------------------------------------------------------------
__global__ void k_prep(const float* __restrict__ x, const float* __restrict__ maa_x,
                       float* __restrict__ xx, bf16_t* __restrict__ Xx) {
  size_t idx = (size_t)blockIdx.x * blockDim.x + threadIdx.x;
  if (idx >= (size_t)RWKV_M * RWKV_C) return;
  int c = (int)(idx % RWKV_C);
  int t = (int)((idx / RWKV_C) % RWKV_T);
  float xv = x[idx];
  float prev = (t == 0) ? 0.f : x[idx - RWKV_C];
  float d = prev - xv;
  xx[idx] = d;
  Xx[idx] = (bf16_t)(xv + d * maa_x[c]);
}

// ---------------------------------------------------------------------------
// Generic WMMA bf16 GEMM: C[M,N] = A[M,K](bf16,row-major) * Bt[N,K](bf16)
// One wave computes a (MT*16) x (NT*16) tile: MT x NT accumulators,
// MT + NT fragment loads feed MT*NT WMMAs per K-step (raises WMMA density
// and A/B reuse; everything streams through the 192MB L2).
// ---------------------------------------------------------------------------
enum { EPI_NONE = 0, EPI_SILU = 1, EPI_TANH_BF16 = 2, EPI_DECAY = 3 };

template <int MT, int NT, int EPI>
__global__ void k_gemm_bf16(const bf16_t* __restrict__ A, const bf16_t* __restrict__ Bt,
                            void* __restrict__ Cout, int M, int N, int K,
                            const float* __restrict__ aux) {
  const int lane = threadIdx.x & 31;
  const int wid  = (int)((blockIdx.x * (size_t)blockDim.x + threadIdx.x) >> 5);
  const int nwt = N / (NT * 16);
  const int tm = wid / nwt;
  const int tn = wid % nwt;
  if (tm >= M / (MT * 16)) return;  // wave-uniform (grids launched exact)

  const bf16_t* arow = A  + (size_t)(tm * MT * 16) * K;
  const bf16_t* brow = Bt + (size_t)(tn * NT * 16) * K;

  v8f acc[MT][NT] = {};
  for (int k0 = 0; k0 < K; k0 += 32) {
    if (k0 + 32 < K) {  // speculative prefetch of next K-step (global_prefetch_b8)
      __builtin_prefetch(arow + k0 + 32, 0, 0);
      __builtin_prefetch(brow + k0 + 32, 0, 0);
    }
    v16bf af[MT], bfv[NT];
#pragma unroll
    for (int i = 0; i < MT; ++i)
      af[i] = frag_load(arow + (size_t)i * 16 * K + k0, K, lane);
#pragma unroll
    for (int j = 0; j < NT; ++j)
      bfv[j] = frag_load(brow + (size_t)j * 16 * K + k0, K, lane);
#pragma unroll
    for (int i = 0; i < MT; ++i)
#pragma unroll
      for (int j = 0; j < NT; ++j)
        acc[i][j] = __builtin_amdgcn_wmma_f32_16x16x32_bf16(
            false, af[i], false, bfv[j], (short)0, acc[i][j], false, false);
  }

  const int cl = lane & 15;
  const int rb = (lane >> 4) << 3;
#pragma unroll
  for (int i = 0; i < MT; ++i) {
#pragma unroll
    for (int j = 0; j < NT; ++j) {
      const int col = tn * NT * 16 + j * 16 + cl;
#pragma unroll
      for (int r2 = 0; r2 < 8; ++r2) {
        const int row = tm * MT * 16 + i * 16 + rb + r2;
        const size_t oidx = (size_t)row * N + col;
        float v = acc[i][j][r2];
        if (EPI == EPI_NONE) {
          ((float*)Cout)[oidx] = v;
        } else if (EPI == EPI_SILU) {
          ((float*)Cout)[oidx] = v / (1.f + __expf(-v));
        } else if (EPI == EPI_TANH_BF16) {
          ((bf16_t*)Cout)[oidx] = (bf16_t)tanhf(v);
        } else {  // EPI_DECAY: decay = exp(-exp(time_decay[col] + v))
          ((float*)Cout)[oidx] = __expf(-__expf(aux[col] + v));
        }
      }
    }
  }
}

// ---------------------------------------------------------------------------
// 5-way mixing: m_f = xxx[:, f*32:(f+1)*32] @ W2[f]  (K=32 -> one WMMA each)
// X_f = bf16(x + xx * (maa_f + m_f)), all 5 outputs from one tile pass.
// ---------------------------------------------------------------------------
__global__ void k_mix5(const bf16_t* __restrict__ xxx, const bf16_t* __restrict__ W2t,
                       const float* __restrict__ x, const float* __restrict__ xx,
                       const float* __restrict__ maa_w, const float* __restrict__ maa_k,
                       const float* __restrict__ maa_v, const float* __restrict__ maa_r,
                       const float* __restrict__ maa_g,
                       bf16_t* __restrict__ Xw, bf16_t* __restrict__ Xk,
                       bf16_t* __restrict__ Xv, bf16_t* __restrict__ Xr,
                       bf16_t* __restrict__ Xg) {
  const int lane = threadIdx.x & 31;
  const int wid  = (int)((blockIdx.x * (size_t)blockDim.x + threadIdx.x) >> 5);
  const int ntiles = RWKV_C >> 4;
  const int tm = wid / ntiles;
  const int tn = wid % ntiles;
  const int col   = tn * 16 + (lane & 15);
  const int rbase = tm * 16 + ((lane >> 4) << 3);

  float xv8[8], xxv8[8];
#pragma unroll
  for (int r2 = 0; r2 < 8; ++r2) {
    size_t idx = (size_t)(rbase + r2) * RWKV_C + col;
    xv8[r2] = x[idx];
    xxv8[r2] = xx[idx];
  }

  const float* maas[5] = {maa_w, maa_k, maa_v, maa_r, maa_g};
  bf16_t* outs[5] = {Xw, Xk, Xv, Xr, Xg};

#pragma unroll
  for (int f = 0; f < 5; ++f) {
    v16bf af = frag_load(xxx + (size_t)(tm * 16) * 160 + f * 32, 160, lane);
    v16bf bb = frag_load(W2t + (size_t)f * RWKV_C * 32 + (size_t)(tn * 16) * 32, 32, lane);
    v8f acc = {};
    acc = __builtin_amdgcn_wmma_f32_16x16x32_bf16(false, af, false, bb,
                                                  (short)0, acc, false, false);
    const float maac = maas[f][col];
#pragma unroll
    for (int r2 = 0; r2 < 8; ++r2) {
      size_t idx = (size_t)(rbase + r2) * RWKV_C + col;
      outs[f][idx] = (bf16_t)(xv8[r2] + xxv8[r2] * (maac + acc[r2]));
    }
  }
}

// ---------------------------------------------------------------------------
// WKV scan: one block per (b,h); 256 threads = 4 j-groups x 64 columns.
// Thread (jg,i) owns S[jg*16 .. jg*16+15][i] in registers (no cross-thread
// state dependency; only the output y is reduced across j-groups).
//   y_i     = sum_j r_j*S[j][i] + v_i * sum_j r_j*u_j*k_j
//   S[j][i] = d_j*S[j][i] + k_j*v_i
// Per-step r/k/v/d staging is double-buffered through LDS with CDNA5
// GLOBAL_LOAD_ASYNC_TO_LDS_B32 (ASYNCcnt), issued one step ahead.
// ---------------------------------------------------------------------------
__global__ void k_wkv(const float* __restrict__ r, const float* __restrict__ k,
                      const float* __restrict__ v, const float* __restrict__ dcy,
                      const float* __restrict__ u, float* __restrict__ y) {
  const int bh = blockIdx.x;
  const int b = bh / RWKV_H;
  const int h = bh % RWKV_H;
  const int tid = threadIdx.x;
  const int i  = tid & 63;
  const int jg = tid >> 6;

  __shared__ float sh[2][4][RWKV_N];
  __shared__ float part[256];

  float s[16];
  float uj[16];
#pragma unroll
  for (int jj = 0; jj < 16; ++jj) {
    s[jj] = 0.f;
    uj[jj] = u[h * RWKV_N + jg * 16 + jj];
  }

  const float* srcp = (jg == 0) ? r : (jg == 1) ? k : (jg == 2) ? v : dcy;
  const size_t base0 = ((size_t)(b * RWKV_T)) * RWKV_C + h * RWKV_N;

  // stage step t into buffer buf (one element per thread)
  auto stage = [&](int buf, int t) {
    const float* src = srcp + base0 + (size_t)t * RWKV_C + i;
#if HAVE_ASYNC_LDS
    __builtin_amdgcn_global_load_async_to_lds_b32(
        AS1_PTR(src), AS3_PTR(&sh[buf][jg][i]), 0, 0);
#else
    sh[buf][jg][i] = *src;
#endif
  };

  stage(0, 0);
  for (int t = 0; t < RWKV_T; ++t) {
    const int cur = t & 1;
#if HAVE_ASYNC_LDS
    __builtin_amdgcn_s_wait_asynccnt(0);   // this wave's async writes landed
#endif
    __syncthreads();                       // all waves' writes visible
    if (t + 1 < RWKV_T) stage(cur ^ 1, t + 1);  // prefetch next step

    const float vi = sh[cur][2][i];
    float t1 = 0.f, t2 = 0.f;
#pragma unroll
    for (int jj = 0; jj < 16; ++jj) {
      const int j = jg * 16 + jj;
      const float rj = sh[cur][0][j], kj = sh[cur][1][j], dj = sh[cur][3][j];
      t1 += rj * s[jj];
      t2 += rj * kj * uj[jj];
      s[jj] = dj * s[jj] + kj * vi;
    }
    part[tid] = t1 + t2 * vi;
    __syncthreads();

    if (jg == 0)
      y[base0 + (size_t)t * RWKV_C + i] =
          part[i] + part[64 + i] + part[128 + i] + part[192 + i];
  }
}

// ---------------------------------------------------------------------------
// Per-head GroupNorm + ln scale/bias + silu-gate multiply -> bf16
// ---------------------------------------------------------------------------
__global__ void k_gn_gate(const float* __restrict__ y, const float* __restrict__ g,
                          const float* __restrict__ lnw, const float* __restrict__ lnb,
                          bf16_t* __restrict__ Ybf) {
  const int grp = blockIdx.x;          // b*T*H + ...
  const int h  = grp % RWKV_H;
  const int bt = grp / RWKV_H;
  const int i  = threadIdx.x;          // 0..63
  const size_t idx = (size_t)bt * RWKV_C + h * RWKV_N + i;
  const float vv = y[idx];

  __shared__ float s1[RWKV_N], s2[RWKV_N];
  s1[i] = vv; s2[i] = vv * vv;
  __syncthreads();
  for (int off = 32; off > 0; off >>= 1) {
    if (i < off) { s1[i] += s1[i + off]; s2[i] += s2[i + off]; }
    __syncthreads();
  }
  const float mean = s1[0] * (1.f / RWKV_N);
  const float var  = s2[0] * (1.f / RWKV_N) - mean * mean;
  const float yn = (vv - mean) * rsqrtf(var + GN_EPS);
  const int c = h * RWKV_N + i;
  Ybf[idx] = (bf16_t)((yn * lnw[c] + lnb[c]) * g[idx]);
}

// ---------------------------------------------------------------------------
// Host-side orchestration
// ---------------------------------------------------------------------------
extern "C" void kernel_launch(void* const* d_in, const int* in_sizes, int n_in,
                              void* d_out, int out_size, void* d_ws, size_t ws_size,
                              hipStream_t stream) {
  const float* x      = (const float*)d_in[0];
  const float* maa_x  = (const float*)d_in[1];
  const float* maa_w  = (const float*)d_in[2];
  const float* maa_k  = (const float*)d_in[3];
  const float* maa_v  = (const float*)d_in[4];
  const float* maa_r  = (const float*)d_in[5];
  const float* maa_g  = (const float*)d_in[6];
  const float* w1     = (const float*)d_in[7];   // [C,160]
  const float* w2     = (const float*)d_in[8];   // [5,32,C]
  const float* tdecay = (const float*)d_in[9];   // [C]
  const float* dw1    = (const float*)d_in[10];  // [C,64]
  const float* dw2    = (const float*)d_in[11];  // [64,C]
  const float* faaaa  = (const float*)d_in[12];  // [H,N]
  const float* Wr     = (const float*)d_in[13];
  const float* Wk     = (const float*)d_in[14];
  const float* Wv     = (const float*)d_in[15];
  const float* Wg     = (const float*)d_in[16];
  const float* Wo     = (const float*)d_in[17];
  const float* lnw    = (const float*)d_in[18];
  const float* lnb    = (const float*)d_in[19];

  const int M = RWKV_M, C = RWKV_C;

  char* ws = (char*)d_ws;
  size_t off = 0;
  auto alloc = [&](size_t bytes) -> char* {
    char* p = ws + off;
    off = (off + bytes + 255) & ~(size_t)255;
    return p;
  };

  float*  xxd   = (float*)alloc((size_t)M * C * 4);
  bf16_t* Xx    = (bf16_t*)alloc((size_t)M * C * 2);
  bf16_t* W1t   = (bf16_t*)alloc((size_t)160 * C * 2);
  bf16_t* xxxb  = (bf16_t*)alloc((size_t)M * 160 * 2);
  bf16_t* W2t   = (bf16_t*)alloc((size_t)5 * C * 32 * 2);
  bf16_t* Wrt   = (bf16_t*)alloc((size_t)C * C * 2);
  bf16_t* Wkt   = (bf16_t*)alloc((size_t)C * C * 2);
  bf16_t* Wvt   = (bf16_t*)alloc((size_t)C * C * 2);
  bf16_t* Wgt   = (bf16_t*)alloc((size_t)C * C * 2);
  bf16_t* Wot   = (bf16_t*)alloc((size_t)C * C * 2);
  bf16_t* D1t   = (bf16_t*)alloc((size_t)64 * C * 2);
  bf16_t* D2t   = (bf16_t*)alloc((size_t)C * 64 * 2);
  bf16_t* Xw    = (bf16_t*)alloc((size_t)M * C * 2);
  bf16_t* Xk    = (bf16_t*)alloc((size_t)M * C * 2);
  bf16_t* Xv    = (bf16_t*)alloc((size_t)M * C * 2);
  bf16_t* Xr    = (bf16_t*)alloc((size_t)M * C * 2);
  bf16_t* Xg    = (bf16_t*)alloc((size_t)M * C * 2);
  float*  rbuf  = (float*)alloc((size_t)M * C * 4);
  float*  kbuf  = (float*)alloc((size_t)M * C * 4);
  float*  vbuf  = (float*)alloc((size_t)M * C * 4);
  float*  gbuf  = (float*)alloc((size_t)M * C * 4);
  float*  dbuf  = (float*)alloc((size_t)M * C * 4);
  bf16_t* hbuf  = (bf16_t*)alloc((size_t)M * 64 * 2);
  float*  ybuf  = (float*)alloc((size_t)M * C * 4);
  bf16_t* Ybf   = (bf16_t*)alloc((size_t)M * C * 2);

  auto tr = [&](const float* S, bf16_t* D, int K, int N) {
    int total = K * N;
    k_transpose_bf16<<<(total + 255) / 256, 256, 0, stream>>>(S, D, K, N);
  };

  // 0) weight conversion (f32 -> bf16, transposed to [N,K] for B-fragments)
  tr(w1, W1t, C, 160);
  for (int f = 0; f < 5; ++f)
    tr(w2 + (size_t)f * 32 * C, W2t + (size_t)f * C * 32, 32, C);
  tr(Wr, Wrt, C, C);
  tr(Wk, Wkt, C, C);
  tr(Wv, Wvt, C, C);
  tr(Wg, Wgt, C, C);
  tr(Wo, Wot, C, C);
  tr(dw1, D1t, C, 64);
  tr(dw2, D2t, 64, C);

  // 1) token shift + base mix
  k_prep<<<(M * C + 255) / 256, 256, 0, stream>>>(x, maa_x, xxd, Xx);

  // waves = (M/(mt*16)) * (N/(nt*16)); 8 waves per 256-thread block
  auto blocks_for = [](int Mm, int Nn, int mt, int nt) {
    return ((Mm / (mt * 16)) * (Nn / (nt * 16))) / 8;
  };

  // 2) xxx = tanh(Xx @ W1)   [M,160] bf16   (32x32 wave tiles: 160 % 64 != 0)
  k_gemm_bf16<2, 2, EPI_TANH_BF16><<<blocks_for(M, 160, 2, 2), 256, 0, stream>>>(
      Xx, W1t, xxxb, M, 160, C, nullptr);

  // 3) five fused K=32 WMMA GEMMs + mixing compose (16x16 tiles, K fits one WMMA)
  k_mix5<<<((M >> 4) * (C >> 4)) / 8, 256, 0, stream>>>(
      xxxb, W2t, x, xxd, maa_w, maa_k, maa_v, maa_r, maa_g, Xw, Xk, Xv, Xr, Xg);

  // 4) big projections (32x64 wave tiles)
  k_gemm_bf16<2, 4, EPI_NONE><<<blocks_for(M, C, 2, 4), 256, 0, stream>>>(
      Xr, Wrt, rbuf, M, C, C, nullptr);
  k_gemm_bf16<2, 4, EPI_NONE><<<blocks_for(M, C, 2, 4), 256, 0, stream>>>(
      Xk, Wkt, kbuf, M, C, C, nullptr);
  k_gemm_bf16<2, 4, EPI_NONE><<<blocks_for(M, C, 2, 4), 256, 0, stream>>>(
      Xv, Wvt, vbuf, M, C, C, nullptr);
  k_gemm_bf16<2, 4, EPI_SILU><<<blocks_for(M, C, 2, 4), 256, 0, stream>>>(
      Xg, Wgt, gbuf, M, C, C, nullptr);

  // 5) decay path: h = tanh(Xw @ dw1) [M,64] bf16; decay = exp(-exp(td + h@dw2))
  k_gemm_bf16<2, 2, EPI_TANH_BF16><<<blocks_for(M, 64, 2, 2), 256, 0, stream>>>(
      Xw, D1t, hbuf, M, 64, C, nullptr);
  k_gemm_bf16<2, 4, EPI_DECAY><<<blocks_for(M, C, 2, 4), 256, 0, stream>>>(
      hbuf, D2t, dbuf, M, C, 64, tdecay);

  // 6) sequential WKV scan (64 blocks, one per (b,h); async double-buffered)
  k_wkv<<<RWKV_B * RWKV_H, 256, 0, stream>>>(rbuf, kbuf, vbuf, dbuf, faaaa, ybuf);

  // 7) GroupNorm per head + gate
  k_gn_gate<<<RWKV_B * RWKV_T * RWKV_H, RWKV_N, 0, stream>>>(ybuf, gbuf, lnw, lnb, Ybf);

  // 8) output projection -> d_out [B,T,C] f32
  k_gemm_bf16<2, 4, EPI_NONE><<<blocks_for(M, C, 2, 4), 256, 0, stream>>>(
      Ybf, Wot, (float*)d_out, M, C, C, nullptr);

  (void)in_sizes; (void)n_in; (void)out_size; (void)ws_size;
}
